// EncoderLayer_71476845740778
// MI455X (gfx1250) — matrix operand order repaired and checked by reading, not compiled, not run on hardware
//
#include <hip/hip_runtime.h>
#include <hip/hip_bf16.h>

#define B_   4
#define S_   4096
#define D_   1024
#define H_   16
#define DK_  64
#define DFF_ 4096
#define MT_  (B_ * S_)   // 16384 rows

typedef __attribute__((ext_vector_type(16))) __bf16 v16bf;
typedef __attribute__((ext_vector_type(8)))  float  v8f;

__device__ __forceinline__ float bf2f(__bf16 v) {
  unsigned short s = __builtin_bit_cast(unsigned short, v);
  unsigned int u = ((unsigned int)s) << 16;
  return __builtin_bit_cast(float, u);
}
__device__ __forceinline__ __bf16 f2bf(float f) {
  unsigned int u = __builtin_bit_cast(unsigned int, f);
  u += 0x7fffu + ((u >> 16) & 1u);           // round-to-nearest-even
  unsigned short s = (unsigned short)(u >> 16);
  return __builtin_bit_cast(__bf16, s);
}

union Frag { v16bf v; uint4 u[2]; };

// issue one async global->LDS 16B copy (CDNA5 path, tracked by ASYNCcnt)
__device__ __forceinline__ void async_cp16(const __bf16* gsrc, __bf16* ldst) {
  unsigned la = (unsigned)(size_t)ldst;              // low 32 bits = LDS offset
  unsigned long long ga = (unsigned long long)(size_t)gsrc;
  asm volatile("global_load_async_to_lds_b128 %0, %1, off"
               :: "v"(la), "v"(ga) : "memory");
}
__device__ __forceinline__ void wait_async0() {
  asm volatile("s_wait_asynccnt 0x0" ::: "memory");
}

// ---------------------------------------------------------------------------
// Weight prep: fp32 [K,N] row-major  ->  bf16 [N,K] row-major (transposed)
// ---------------------------------------------------------------------------
__global__ __launch_bounds__(256) void transpose_cvt(
    const float* __restrict__ in, __bf16* __restrict__ out, int K, int N) {
  __shared__ float t[32][33];
  const int tid = threadIdx.x;
  const int tx = tid & 31, ty = tid >> 5;
  const int n0 = blockIdx.x * 32, k0 = blockIdx.y * 32;
#pragma unroll
  for (int j = 0; j < 4; ++j)
    t[ty + j * 8][tx] = in[(size_t)(k0 + ty + j * 8) * N + n0 + tx];
  __syncthreads();
#pragma unroll
  for (int j = 0; j < 4; ++j)
    out[(size_t)(n0 + ty + j * 8) * K + k0 + tx] = f2bf(t[tx][ty + j * 8]);
}

// ---------------------------------------------------------------------------
// LayerNorm: fp32 [rows, 1024] -> bf16 [rows, 1024]; one block per row
// ---------------------------------------------------------------------------
__global__ __launch_bounds__(256) void ln_bf16(
    const float* __restrict__ x, const float* __restrict__ g,
    const float* __restrict__ b, __bf16* __restrict__ out) {
  __shared__ float s1[256], s2[256];
  const int row = blockIdx.x, tid = threadIdx.x;
  const float* xr = x + (size_t)row * D_;
  float v[4], a = 0.f, q = 0.f;
#pragma unroll
  for (int i = 0; i < 4; ++i) { v[i] = xr[tid + i * 256]; a += v[i]; q += v[i] * v[i]; }
  s1[tid] = a; s2[tid] = q;
  __syncthreads();
  for (int off = 128; off > 0; off >>= 1) {
    if (tid < off) { s1[tid] += s1[tid + off]; s2[tid] += s2[tid + off]; }
    __syncthreads();
  }
  const float mean = s1[0] * (1.f / D_);
  const float var  = s2[0] * (1.f / D_) - mean * mean;
  const float inv  = rsqrtf(var + 1e-5f);
  __bf16* orow = out + (size_t)row * D_;
#pragma unroll
  for (int i = 0; i < 4; ++i) {
    const int c = tid + i * 256;
    orow[c] = f2bf((v[i] - mean) * inv * g[c] + b[c]);
  }
}

// ---------------------------------------------------------------------------
// WMMA GEMM, 128x128 block tile, 8 waves (4x2), wave tile 32x64, K-step 32.
// Double-buffered LDS filled by GLOBAL_LOAD_ASYNC_TO_LDS_B128 (ASYNCcnt),
// one barrier per K-step. All A/B fragments materialized before the WMMA
// block so the scheduler can batch ds_loads and stagger dscnt waits.
// A: bf16 [M,K] row-major.  W1/W2: bf16 [N,K] (pre-transposed).
// EPI 0: out_bf16 = silu(A*W1+b1) * (A*W2+b2)                 (SwiGLU)
// EPI 1: EPI0 then elu(.)+1, optional per-row mask            (phi features)
// EPI 2: out_f32 = A*W1 + b1 + residual                       (proj + residual)
// ---------------------------------------------------------------------------
template <int EPI>
__global__ __launch_bounds__(256) void gemm_wmma(
    const __bf16* __restrict__ A, const __bf16* __restrict__ W1,
    const __bf16* __restrict__ W2, const float* __restrict__ b1,
    const float* __restrict__ b2, const float* __restrict__ mask,
    const float* __restrict__ residual, void* __restrict__ outp,
    int M, int N, int K) {
  constexpr bool DUAL = (EPI < 2);
  __shared__ __bf16 lds_a[2][128][40];               // 80B row stride, 16B-aligned runs
  __shared__ __bf16 lds_w1[2][128][40];
  __shared__ __bf16 lds_w2[DUAL ? 2 : 1][128][40];

  const int tid = threadIdx.x;
  const int m0 = blockIdx.y * 128, n0 = blockIdx.x * 128;
  // staging: 512 x 16B chunks per 8KB tile; thread does rows r0 and r0+64, quarter q0
  const int r0 = tid >> 2, q0 = (tid & 3) * 8;
  const int wave = tid >> 5, lane = tid & 31;
  const int wm = (wave & 3) * 32, wn = (wave >> 2) * 64;
  const int lr = lane & 15, lh = lane >> 4;

  auto stage = [&](int bi, int k0) {
#pragma unroll
    for (int t = 0; t < 2; ++t) {
      const int rr = r0 + t * 64;
      async_cp16(A  + (size_t)(m0 + rr) * K + k0 + q0, &lds_a[bi][rr][q0]);
      async_cp16(W1 + (size_t)(n0 + rr) * K + k0 + q0, &lds_w1[bi][rr][q0]);
      if constexpr (DUAL)
        async_cp16(W2 + (size_t)(n0 + rr) * K + k0 + q0, &lds_w2[bi][rr][q0]);
    }
  };

  v8f accU[2][4], accG[2][4];
#pragma unroll
  for (int i = 0; i < 2; ++i)
#pragma unroll
    for (int j = 0; j < 4; ++j)
#pragma unroll
      for (int l = 0; l < 8; ++l) { accU[i][j][l] = 0.f; accG[i][j][l] = 0.f; }

  const int KT = K >> 5;
  stage(0, 0);
  for (int kt = 0; kt < KT; ++kt) {
    wait_async0();          // this wave's tile-kt async loads have landed in LDS
    __syncthreads();        // publish tile kt; all waves done reading buffer (kt+1)&1
    if (kt + 1 < KT) stage((kt + 1) & 1, (kt + 1) << 5);
    const int bi = kt & 1;

    // Materialize all fragments first: lets the scheduler issue all ds_loads
    // back-to-back and stagger the dscnt waits across the WMMA drain.
    // A fragments: lanes 0-15 hold K 0-15; lanes 16-31 hold K 16-31
    Frag fa[2];
#pragma unroll
    for (int mi = 0; mi < 2; ++mi) {
      const int r = wm + mi * 16 + lr;
      fa[mi].u[0] = *(const uint4*)&lds_a[bi][r][lh * 16];
      fa[mi].u[1] = *(const uint4*)&lds_a[bi][r][lh * 16 + 8];
    }
    Frag fb1[4], fb2[4];
#pragma unroll
    for (int ni = 0; ni < 4; ++ni) {
      const int c = wn + ni * 16 + lr;   // B: lane holds column lr, K-run lh*16..+15
      fb1[ni].u[0] = *(const uint4*)&lds_w1[bi][c][lh * 16];
      fb1[ni].u[1] = *(const uint4*)&lds_w1[bi][c][lh * 16 + 8];
      if constexpr (DUAL) {
        fb2[ni].u[0] = *(const uint4*)&lds_w2[bi][c][lh * 16];
        fb2[ni].u[1] = *(const uint4*)&lds_w2[bi][c][lh * 16 + 8];
      }
    }
#pragma unroll
    for (int ni = 0; ni < 4; ++ni) {
#pragma unroll
      for (int mi = 0; mi < 2; ++mi)
        accU[mi][ni] = __builtin_amdgcn_wmma_f32_16x16x32_bf16(
            false, fa[mi].v, false, fb1[ni].v, (short)0, accU[mi][ni], false, false);
      if constexpr (DUAL) {
#pragma unroll
        for (int mi = 0; mi < 2; ++mi)
          accG[mi][ni] = __builtin_amdgcn_wmma_f32_16x16x32_bf16(
              false, fa[mi].v, false, fb2[ni].v, (short)0, accG[mi][ni], false, false);
      }
    }
  }

  // Epilogue. C layout: col = lr, row = lh*8 + i (per 16x16 tile)
#pragma unroll
  for (int ni = 0; ni < 4; ++ni) {
    const int col = n0 + wn + ni * 16 + lr;
    const float bv1 = b1[col];
    const float bv2 = DUAL ? b2[col] : 0.f;
#pragma unroll
    for (int mi = 0; mi < 2; ++mi) {
#pragma unroll
      for (int i = 0; i < 8; ++i) {
        const int row = m0 + wm + mi * 16 + lh * 8 + i;
        if constexpr (EPI == 2) {
          float r = accU[mi][ni][i] + bv1 + residual[(size_t)row * N + col];
          ((float*)outp)[(size_t)row * N + col] = r;
        } else {
          const float u = accU[mi][ni][i] + bv1;
          const float g = accG[mi][ni][i] + bv2;
          float o = (u / (1.f + __expf(-u))) * g;   // silu(u)*g
          if constexpr (EPI == 1) {
            o = (o > 0.f) ? (o + 1.f) : __expf(o);  // elu(o)+1
            if (mask) o *= mask[row];               // row == b*S+s
          }
          ((__bf16*)outp)[(size_t)row * N + col] = f2bf(o);
        }
      }
    }
  }
}

// ---------------------------------------------------------------------------
// kv[b,h,d,e] = sum_s phi_k[b,s,h*64+d] * v[b,s,h*64+e];  ksum[b,h,d] = sum_s phi_k
// One block per (b,h); thread owns 16 d values for one e.
// ---------------------------------------------------------------------------
__global__ __launch_bounds__(256) void kv_reduce(
    const __bf16* __restrict__ kf, const __bf16* __restrict__ vf,
    float* __restrict__ kv, float* __restrict__ ksum) {
  __shared__ __bf16 sk[32][64], sv[32][64];
  const int tid = threadIdx.x;
  const int b = blockIdx.x >> 4, h = blockIdx.x & 15;
  const int e = tid & 63, dg = tid >> 6;
  const int sr = tid >> 3, sc = (tid & 7) * 8;
  float acc[16];
#pragma unroll
  for (int i = 0; i < 16; ++i) acc[i] = 0.f;
  float ks = 0.f;
  const size_t base = (size_t)b * S_ * D_ + h * DK_;
  for (int s0 = 0; s0 < S_; s0 += 32) {
    __syncthreads();
    *(uint4*)&sk[sr][sc] = *(const uint4*)(kf + base + (size_t)(s0 + sr) * D_ + sc);
    *(uint4*)&sv[sr][sc] = *(const uint4*)(vf + base + (size_t)(s0 + sr) * D_ + sc);
    __syncthreads();
#pragma unroll 4
    for (int s = 0; s < 32; ++s) {
      const float vv = bf2f(sv[s][e]);
#pragma unroll
      for (int i = 0; i < 16; ++i) acc[i] += bf2f(sk[s][dg * 16 + i]) * vv;
    }
    if (tid < 64) {
#pragma unroll 4
      for (int s = 0; s < 32; ++s) ks += bf2f(sk[s][tid]);
    }
  }
  float* kvo = kv + (size_t)blockIdx.x * 64 * 64;
#pragma unroll
  for (int i = 0; i < 16; ++i) kvo[(dg * 16 + i) * 64 + e] = acc[i];
  if (tid < 64) ksum[blockIdx.x * 64 + tid] = ks;
}

// ---------------------------------------------------------------------------
// attn[b,s,h*64+e] = (phi_q . kv[:,e]) / (phi_q . ksum + eps)
// grid (S/256, B*H); kv staged in LDS (bank-conflict-free on e)
// ---------------------------------------------------------------------------
__global__ __launch_bounds__(256) void attn_apply(
    const __bf16* __restrict__ qf, const float* __restrict__ kv,
    const float* __restrict__ ksum, __bf16* __restrict__ attn) {
  __shared__ float skv[64][64];
  __shared__ float sks[64];
  __shared__ __bf16 sq[16][64];
  const int tid = threadIdx.x;
  const int bh = blockIdx.y;
  const int b = bh >> 4, h = bh & 15;
  for (int i = tid; i < 4096; i += 256) skv[i >> 6][i & 63] = kv[(size_t)bh * 4096 + i];
  if (tid < 64) sks[tid] = ksum[bh * 64 + tid];
  const int e = tid & 63, rg = tid >> 6;
  const size_t base = (size_t)b * S_ * D_ + h * DK_;
  for (int o = 0; o < 16; ++o) {
    const int s0 = blockIdx.x * 256 + o * 16;
    __syncthreads();
    if (tid < 128) {
      const int r = tid >> 3, c = (tid & 7) * 8;
      *(uint4*)&sq[r][c] = *(const uint4*)(qf + base + (size_t)(s0 + r) * D_ + c);
    }
    __syncthreads();
#pragma unroll
    for (int p = 0; p < 4; ++p) {
      const int r = p * 4 + rg;
      float num = 0.f, den = 0.f;
#pragma unroll 8
      for (int d = 0; d < 64; ++d) {
        const float q = bf2f(sq[r][d]);
        num += q * skv[d][e];
        den += q * sks[d];
      }
      attn[base + (size_t)(s0 + r) * D_ + e] = f2bf(num / (den + 1e-6f));
    }
  }
}

// ---------------------------------------------------------------------------
extern "C" void kernel_launch(void* const* d_in, const int* in_sizes, int n_in,
                              void* d_out, int out_size, void* d_ws, size_t ws_size,
                              hipStream_t stream) {
  const float* x     = (const float*)d_in[0];
  const float* mask  = (const float*)d_in[1];
  const float* ln1_g = (const float*)d_in[2];
  const float* ln1_b = (const float*)d_in[3];
  const float* ln2_g = (const float*)d_in[4];
  const float* ln2_b = (const float*)d_in[5];
  const float* q_w1  = (const float*)d_in[6];  const float* q_b1 = (const float*)d_in[7];
  const float* q_w2  = (const float*)d_in[8];  const float* q_b2 = (const float*)d_in[9];
  const float* k_w1  = (const float*)d_in[10]; const float* k_b1 = (const float*)d_in[11];
  const float* k_w2  = (const float*)d_in[12]; const float* k_b2 = (const float*)d_in[13];
  const float* v_w1  = (const float*)d_in[14]; const float* v_b1 = (const float*)d_in[15];
  const float* v_w2  = (const float*)d_in[16]; const float* v_b2 = (const float*)d_in[17];
  const float* out_w = (const float*)d_in[18]; const float* out_b = (const float*)d_in[19];
  const float* ff_w1 = (const float*)d_in[20]; const float* ff_b1 = (const float*)d_in[21];
  const float* ff_w2 = (const float*)d_in[22]; const float* ff_b2 = (const float*)d_in[23];
  const float* ff_w3 = (const float*)d_in[24]; const float* ff_b3 = (const float*)d_in[25];

  char* ws = (char*)d_ws;
  size_t off = 0;
  auto alloc = [&](size_t bytes) -> void* {
    void* p = ws + off;
    off += (bytes + 255) & ~(size_t)255;
    return p;
  };
  const size_t DD = (size_t)D_ * D_, DF = (size_t)D_ * DFF_, MD = (size_t)MT_ * D_;
  __bf16* wq1t = (__bf16*)alloc(DD * 2);  __bf16* wq2t = (__bf16*)alloc(DD * 2);
  __bf16* wk1t = (__bf16*)alloc(DD * 2);  __bf16* wk2t = (__bf16*)alloc(DD * 2);
  __bf16* wv1t = (__bf16*)alloc(DD * 2);  __bf16* wv2t = (__bf16*)alloc(DD * 2);
  __bf16* wot  = (__bf16*)alloc(DD * 2);
  __bf16* wf1t = (__bf16*)alloc(DF * 2);  __bf16* wf3t = (__bf16*)alloc(DF * 2);
  __bf16* wf2t = (__bf16*)alloc(DF * 2);
  __bf16* x2   = (__bf16*)alloc(MD * 2);
  __bf16* qf   = (__bf16*)alloc(MD * 2);
  __bf16* kf   = (__bf16*)alloc(MD * 2);
  __bf16* vf   = (__bf16*)alloc(MD * 2);
  __bf16* attn = (__bf16*)alloc(MD * 2);
  __bf16* x3   = (__bf16*)alloc(MD * 2);
  float*  kvb  = (float*)alloc((size_t)B_ * H_ * DK_ * DK_ * 4);
  float*  ksum = (float*)alloc((size_t)B_ * H_ * DK_ * 4);
  float*  xmid = (float*)alloc(MD * 4);
  __bf16* ffh  = (__bf16*)alloc((size_t)MT_ * DFF_ * 2);
  (void)ws_size; (void)in_sizes; (void)n_in; (void)out_size;

  // weight prep: transpose + f32->bf16
  auto T = [&](const float* in, __bf16* out, int K, int N) {
    transpose_cvt<<<dim3(N / 32, K / 32), 256, 0, stream>>>(in, out, K, N);
  };
  T(q_w1, wq1t, D_, D_);   T(q_w2, wq2t, D_, D_);
  T(k_w1, wk1t, D_, D_);   T(k_w2, wk2t, D_, D_);
  T(v_w1, wv1t, D_, D_);   T(v_w2, wv2t, D_, D_);
  T(out_w, wot, D_, D_);
  T(ff_w1, wf1t, D_, DFF_); T(ff_w3, wf3t, D_, DFF_);
  T(ff_w2, wf2t, DFF_, D_);

  // LN1
  ln_bf16<<<MT_, 256, 0, stream>>>(x, ln1_g, ln1_b, x2);

  // phi_q, phi_k (masked), v  — fused dual-GEMM SwiGLU projections
  dim3 gD(D_ / 128, MT_ / 128);
  gemm_wmma<1><<<gD, 256, 0, stream>>>(x2, wq1t, wq2t, q_b1, q_b2, nullptr, nullptr, qf, MT_, D_, D_);
  gemm_wmma<1><<<gD, 256, 0, stream>>>(x2, wk1t, wk2t, k_b1, k_b2, mask,    nullptr, kf, MT_, D_, D_);
  gemm_wmma<0><<<gD, 256, 0, stream>>>(x2, wv1t, wv2t, v_b1, v_b2, nullptr, nullptr, vf, MT_, D_, D_);

  // linear attention state + application
  kv_reduce<<<B_ * H_, 256, 0, stream>>>(kf, vf, kvb, ksum);
  attn_apply<<<dim3(S_ / 256, B_ * H_), 256, 0, stream>>>(qf, kvb, ksum, attn);

  // out projection + residual (fp32)
  gemm_wmma<2><<<gD, 256, 0, stream>>>(attn, wot, nullptr, out_b, nullptr, nullptr, x, xmid, MT_, D_, D_);

  // LN2 + SwiGLU FFN + residual
  ln_bf16<<<MT_, 256, 0, stream>>>(xmid, ln2_g, ln2_b, x3);
  gemm_wmma<0><<<dim3(DFF_ / 128, MT_ / 128), 256, 0, stream>>>(
      x3, wf1t, wf3t, ff_b1, ff_b3, nullptr, nullptr, ffh, MT_, DFF_, D_);
  gemm_wmma<2><<<gD, 256, 0, stream>>>(
      ffh, wf2t, nullptr, ff_b2, nullptr, nullptr, xmid, (float*)d_out, MT_, D_, DFF_);
}
